// UpdDenseNet_57372173140205
// MI455X (gfx1250) — matrix operand order repaired and checked by reading, not compiled, not run on hardware
//
#include <hip/hip_runtime.h>
#include <hip/hip_bf16.h>

#define N_ASSETS 48
#define COV_ROWS 50
#define BATCH 512
#define N_FEAT 24576            // 8*64*48
#define N_COV_OUT 2400          // 48*50
#define MAX_W 1.5f
#define BN_EPS 1e-5f
#define PGD_ITERS 200
#define BISECT_ITERS 50

typedef __attribute__((ext_vector_type(16))) _Float16 v16h;
typedef __attribute__((ext_vector_type(8)))  _Float16 h8;
typedef __attribute__((ext_vector_type(4)))  _Float16 h4;
typedef __attribute__((ext_vector_type(8)))  float    v8f;

// ---------------- BatchNorm statistics: per-feature scale/shift ----------------
__global__ __launch_bounds__(256) void bn_stats_kernel(
    const float* __restrict__ x, const float* __restrict__ gamma,
    const float* __restrict__ beta, float* __restrict__ scale,
    float* __restrict__ shift)
{
  int f = blockIdx.x * 256 + threadIdx.x;
  if (f >= N_FEAT) return;
  float s = 0.f, ss = 0.f;
  const float* p = x + f;
  #pragma unroll 8
  for (int r = 0; r < BATCH; ++r) {
    float v = p[(size_t)r * N_FEAT];
    s += v;
    ss = fmaf(v, v, ss);
  }
  float mean = s * (1.f / BATCH);
  float var  = ss * (1.f / BATCH) - mean * mean;   // biased, as jnp.var
  float sc   = rsqrtf(var + BN_EPS) * gamma[f];
  scale[f] = sc;
  shift[f] = fmaf(-mean, sc, beta[f]);
}

// ---------------- Apply BN and convert to f16 (WMMA A operand) ----------------
__global__ __launch_bounds__(256) void bn_apply_kernel(
    const float* __restrict__ x, const float* __restrict__ scale,
    const float* __restrict__ shift, _Float16* __restrict__ xn)
{
  int i4 = (blockIdx.x * 256 + threadIdx.x) * 4;
  if (i4 >= BATCH * N_FEAT) return;
  int f = i4 % N_FEAT;                 // N_FEAT % 4 == 0 -> stays in-row
  float4 xv = *(const float4*)(x + i4);
  float4 sc = *(const float4*)(scale + f);
  float4 sh = *(const float4*)(shift + f);
  h4 o;
  o[0] = (_Float16)fmaf(xv.x, sc.x, sh.x);
  o[1] = (_Float16)fmaf(xv.y, sc.y, sh.y);
  o[2] = (_Float16)fmaf(xv.z, sc.z, sh.z);
  o[3] = (_Float16)fmaf(xv.w, sc.w, sh.w);
  *(h4*)(xn + i4) = o;
}

// ---------------- One-shot fp32 -> f16 weight conversion (bandwidth bound) ----
__global__ __launch_bounds__(256) void wcvt_kernel(
    const float* __restrict__ w, _Float16* __restrict__ o, int n)
{
  int i8 = (blockIdx.x * 256 + threadIdx.x) * 8;
  if (i8 >= n) return;
  float4 q0 = *(const float4*)(w + i8);
  float4 q1 = *(const float4*)(w + i8 + 4);
  h8 r;
  r[0] = (_Float16)q0.x; r[1] = (_Float16)q0.y;
  r[2] = (_Float16)q0.z; r[3] = (_Float16)q0.w;
  r[4] = (_Float16)q1.x; r[5] = (_Float16)q1.y;
  r[6] = (_Float16)q1.z; r[7] = (_Float16)q1.w;
  *(h8*)(o + i8) = r;
}

// ---------------- WMMA GEMM: C = act(A @ W^T + bias) ----------------
// A: (M,K) f16 row-major.  W: (N,K) row-major; WF16 ? f16 : fp32 (converted
// in-register).  8 waves/block; wave tile 32(M) x 64(N): 2 A-frags x 4 B-frags,
// 8 WMMAs per K-step.  ACT: 0 = relu, 1 = tanh.
template<int WAVES_M, int WAVES_N, int ACT, bool WF16>
__global__ __launch_bounds__(256) void gemm_wmma_kernel(
    const _Float16* __restrict__ A, const void* __restrict__ Wv,
    const float* __restrict__ bias, float* __restrict__ C,
    int M, int N, int K)
{
  const int lane  = threadIdx.x & 31;
  const int wave  = threadIdx.x >> 5;
  const int waveM = wave / WAVES_N;
  const int waveN = wave % WAVES_N;
  const int lrow  = lane & 15;
  const int khalf = lane >> 4;

  const int rowBase = blockIdx.y * (WAVES_M * 32) + waveM * 32;
  const int colBase = blockIdx.x * (WAVES_N * 64) + waveN * 64;

  int colLoad[4];
  #pragma unroll
  for (int f = 0; f < 4; ++f) {
    int c = colBase + f * 16 + lrow;
    colLoad[f] = (c < N) ? c : (N - 1);       // clamp loads; stores are guarded
  }

  v8f acc[2][4] = {};
  const _Float16* arow0 = A + (size_t)(rowBase + lrow) * K;
  const _Float16* arow1 = A + (size_t)(rowBase + 16 + lrow) * K;

  for (int kk = 0; kk < K; kk += 32) {
    // A fragments (16x32 f16): per-lane contiguous 16-K run at ka
    const int ka = kk + khalf * 16;
    v16h af0, af1;
    ((h8*)&af0)[0] = *(const h8*)(arow0 + ka);
    ((h8*)&af0)[1] = *(const h8*)(arow0 + ka + 8);
    ((h8*)&af1)[0] = *(const h8*)(arow1 + ka);
    ((h8*)&af1)[1] = *(const h8*)(arow1 + ka + 8);

    // B fragments (32x16 f16): per-lane contiguous 16-K run at kb
    const int kb = kk + khalf * 16;
    v16h bf[4];
    #pragma unroll
    for (int f = 0; f < 4; ++f) {
      if (WF16) {
        const _Float16* wr = (const _Float16*)Wv + (size_t)colLoad[f] * K + kb;
        ((h8*)&bf[f])[0] = *(const h8*)(wr);
        ((h8*)&bf[f])[1] = *(const h8*)(wr + 8);
      } else {
        const float* wr = (const float*)Wv + (size_t)colLoad[f] * K + kb;
        float4 q0 = ((const float4*)wr)[0];
        float4 q1 = ((const float4*)wr)[1];
        float4 q2 = ((const float4*)wr)[2];
        float4 q3 = ((const float4*)wr)[3];
        bf[f][0]  = (_Float16)q0.x; bf[f][1]  = (_Float16)q0.y;
        bf[f][2]  = (_Float16)q0.z; bf[f][3]  = (_Float16)q0.w;
        bf[f][4]  = (_Float16)q1.x; bf[f][5]  = (_Float16)q1.y;
        bf[f][6]  = (_Float16)q1.z; bf[f][7]  = (_Float16)q1.w;
        bf[f][8]  = (_Float16)q2.x; bf[f][9]  = (_Float16)q2.y;
        bf[f][10] = (_Float16)q2.z; bf[f][11] = (_Float16)q2.w;
        bf[f][12] = (_Float16)q3.x; bf[f][13] = (_Float16)q3.y;
        bf[f][14] = (_Float16)q3.z; bf[f][15] = (_Float16)q3.w;
      }
    }

    #pragma unroll
    for (int f = 0; f < 4; ++f) {
      acc[0][f] = __builtin_amdgcn_wmma_f32_16x16x32_f16(
          false, af0, false, bf[f], (short)0, acc[0][f], false, false);
      acc[1][f] = __builtin_amdgcn_wmma_f32_16x16x32_f16(
          false, af1, false, bf[f], (short)0, acc[1][f], false, false);
    }
  }

  // Epilogue: C/D layout -> VGPR r holds (m = r + 8*khalf, n = lrow)
  #pragma unroll
  for (int m = 0; m < 2; ++m) {
    #pragma unroll
    for (int f = 0; f < 4; ++f) {
      int col = colBase + f * 16 + lrow;
      if (col < N) {
        float bv = bias[col];
        #pragma unroll
        for (int r = 0; r < 8; ++r) {
          int row = rowBase + m * 16 + khalf * 8 + r;
          float v = acc[m][f][r] + bv;
          v = (ACT == 0) ? fmaxf(v, 0.f) : tanhf(v);
          C[(size_t)row * N + col] = v;
        }
      }
    }
  }
}

// ---------------- wave32 reductions ----------------
__device__ __forceinline__ float wsum(float v) {
  #pragma unroll
  for (int i = 16; i > 0; i >>= 1) v += __shfl_xor(v, i, 32);
  return v;
}
__device__ __forceinline__ float wmaxr(float v) {
  #pragma unroll
  for (int i = 16; i > 0; i >>= 1) v = fmaxf(v, __shfl_xor(v, i, 32));
  return v;
}
__device__ __forceinline__ float wminr(float v) {
  #pragma unroll
  for (int i = 16; i > 0; i >>= 1) v = fminf(v, __shfl_xor(v, i, 32));
  return v;
}

// ---------------- Fused cov -> Q -> projected-gradient QP (one wave / sample) --
// Uses S^T S == cov (S is the symmetric PSD sqrt of a Gram matrix), so eigh is
// eliminated: Q = 2*(g^2*cov + |alpha|*I).
__global__ __launch_bounds__(32) void pgd_kernel(
    const float* __restrict__ Y, const float* __restrict__ RETS,
    const float* __restrict__ gamma_sqrt, const float* __restrict__ alpha,
    float* __restrict__ out)
{
  __shared__ float s_y[COV_ROWS * N_ASSETS];
  __shared__ float s_mean[N_ASSETS];
  __shared__ float s_Q[N_ASSETS * 49];      // row-padded: 49 odd -> conflict-free
  __shared__ float s_w[N_ASSETS];

  const int b    = blockIdx.x;
  const int lane = threadIdx.x;
  const float g2 = gamma_sqrt[0] * gamma_sqrt[0];
  const float al = fabsf(alpha[0]);

  const float* y = Y + (size_t)b * (COV_ROWS * N_ASSETS);
  for (int i = lane; i < COV_ROWS * N_ASSETS; i += 32) s_y[i] = y[i];
  __syncthreads();

  const int a0 = lane;
  const int a1 = (lane < 16) ? lane + 32 : lane;   // lanes>=16 duplicate (unused)

  // column means
  float m0 = 0.f, m1 = 0.f;
  for (int r = 0; r < COV_ROWS; ++r) {
    m0 += s_y[r * N_ASSETS + a0];
    m1 += s_y[r * N_ASSETS + a1];
  }
  m0 *= (1.f / COV_ROWS); m1 *= (1.f / COV_ROWS);
  s_mean[a0] = m0;
  if (lane < 16) s_mean[a1] = m1;
  __syncthreads();

  // Q = 2*(g2*cov + al*I), cov[a][b] = (sum_r y_ra y_rb - R*ma*mb)/(R-1)
  for (int bc = 0; bc < N_ASSETS; ++bc) {
    float mb = s_mean[bc];
    float d0 = 0.f, d1 = 0.f;
    for (int r = 0; r < COV_ROWS; ++r) {
      float yb = s_y[r * N_ASSETS + bc];
      d0 = fmaf(s_y[r * N_ASSETS + a0], yb, d0);
      d1 = fmaf(s_y[r * N_ASSETS + a1], yb, d1);
    }
    float cov0 = (d0 - COV_ROWS * m0 * mb) * (1.f / (COV_ROWS - 1));
    float cov1 = (d1 - COV_ROWS * m1 * mb) * (1.f / (COV_ROWS - 1));
    s_Q[a0 * 49 + bc] = 2.f * (g2 * cov0 + ((a0 == bc) ? al : 0.f));
    if (lane < 16) s_Q[a1 * 49 + bc] = 2.f * (g2 * cov1 + ((a1 == bc) ? al : 0.f));
  }
  __syncthreads();

  // step = 1 / max_i sum_j |Q_ij|
  float rs0 = 0.f, rs1 = 0.f;
  for (int j = 0; j < N_ASSETS; ++j) {
    rs0 += fabsf(s_Q[a0 * 49 + j]);
    rs1 += fabsf(s_Q[a1 * 49 + j]);
  }
  float step = 1.f / wmaxr(fmaxf(rs0, (lane < 16) ? rs1 : 0.f));

  float c0 = RETS[b * N_ASSETS + a0];
  float c1 = RETS[b * N_ASSETS + a1];

  float w0 = 1.f / N_ASSETS, w1 = 1.f / N_ASSETS;
  s_w[a0] = w0;
  if (lane < 16) s_w[a1] = w1;
  __syncthreads();

  for (int it = 0; it < PGD_ITERS; ++it) {
    float g0 = 0.f, g1 = 0.f;
    for (int j = 0; j < N_ASSETS; ++j) {
      float wj = s_w[j];                       // broadcast read
      g0 = fmaf(s_Q[a0 * 49 + j], wj, g0);
      g1 = fmaf(s_Q[a1 * 49 + j], wj, g1);
    }
    float v0 = w0 - step * (g0 - c0);
    float v1 = w1 - step * (g1 - c1);

    // bisection projection onto {sum w = 1, |w_i| <= MAX_W}
    float lo = wminr(fminf(v0, (lane < 16) ? v1 : v0)) - MAX_W - 1.f;
    float hi = wmaxr(fmaxf(v0, (lane < 16) ? v1 : v0)) + MAX_W + 1.f;
    for (int bi = 0; bi < BISECT_ITERS; ++bi) {
      float mid = 0.5f * (lo + hi);
      float t0 = fminf(fmaxf(v0 - mid, -MAX_W), MAX_W);
      float t1 = fminf(fmaxf(v1 - mid, -MAX_W), MAX_W);
      float s = wsum(t0 + ((lane < 16) ? t1 : 0.f));
      bool gt = s > 1.f;
      lo = gt ? mid : lo;
      hi = gt ? hi : mid;
    }
    float lam = 0.5f * (lo + hi);
    w0 = fminf(fmaxf(v0 - lam, -MAX_W), MAX_W);
    w1 = fminf(fmaxf(v1 - lam, -MAX_W), MAX_W);
    s_w[a0] = w0;
    if (lane < 16) s_w[a1] = w1;
    __syncthreads();
  }

  out[b * N_ASSETS + a0] = w0;
  if (lane < 16) out[b * N_ASSETS + a1] = w1;
}

// ---------------- launch ----------------
extern "C" void kernel_launch(void* const* d_in, const int* in_sizes, int n_in,
                              void* d_out, int out_size, void* d_ws, size_t ws_size,
                              hipStream_t stream)
{
  const float* x     = (const float*)d_in[0];
  const float* gamma = (const float*)d_in[1];
  const float* beta  = (const float*)d_in[2];
  const float* Wcov  = (const float*)d_in[3];
  const float* bcov  = (const float*)d_in[4];
  const float* Wlin  = (const float*)d_in[5];
  const float* blin  = (const float*)d_in[6];
  const float* gsq   = (const float*)d_in[7];
  const float* alp   = (const float*)d_in[8];
  float* out = (float*)d_out;

  // workspace layout: base ~30.4 MB, + ~120.3 MB for f16 weight copies
  float*    scale = (float*)d_ws;                          // N_FEAT
  float*    shift = scale + N_FEAT;                        // N_FEAT
  _Float16* xn    = (_Float16*)(shift + N_FEAT);           // BATCH*N_FEAT f16
  float*    y     = (float*)(xn + (size_t)BATCH * N_FEAT); // BATCH*2400
  float*    rets  = y + (size_t)BATCH * N_COV_OUT;         // BATCH*48
  _Float16* wcov16 = (_Float16*)(rets + (size_t)BATCH * N_ASSETS);
  _Float16* wlin16 = wcov16 + (size_t)N_COV_OUT * N_FEAT;

  const size_t base_bytes =
      (size_t)(2 * N_FEAT) * 4 + (size_t)BATCH * N_FEAT * 2 +
      (size_t)BATCH * N_COV_OUT * 4 + (size_t)BATCH * N_ASSETS * 4;
  const size_t f16_bytes =
      base_bytes + ((size_t)N_COV_OUT + N_ASSETS) * N_FEAT * 2;

  bn_stats_kernel<<<(N_FEAT + 255) / 256, 256, 0, stream>>>(x, gamma, beta, scale, shift);
  bn_apply_kernel<<<(BATCH * N_FEAT / 4 + 255) / 256, 256, 0, stream>>>(x, scale, shift, xn);

  if (ws_size >= f16_bytes) {
    // Pre-convert weights once (HBM-bound, ~15 us) so the GEMM inner loop is
    // pure load_b128 + wmma and the XDL pipe is not issue-starved by cvts.
    const int ncov = N_COV_OUT * N_FEAT, nlin = N_ASSETS * N_FEAT;
    wcvt_kernel<<<(ncov / 8 + 255) / 256, 256, 0, stream>>>(Wcov, wcov16, ncov);
    wcvt_kernel<<<(nlin / 8 + 255) / 256, 256, 0, stream>>>(Wlin, wlin16, nlin);
    gemm_wmma_kernel<2, 4, 0, true><<<dim3((N_COV_OUT + 255) / 256, BATCH / 64), 256, 0, stream>>>(
        xn, wcov16, bcov, y, BATCH, N_COV_OUT, N_FEAT);
    gemm_wmma_kernel<8, 1, 1, true><<<dim3(1, BATCH / 256), 256, 0, stream>>>(
        xn, wlin16, blin, rets, BATCH, N_ASSETS, N_FEAT);
  } else {
    gemm_wmma_kernel<2, 4, 0, false><<<dim3((N_COV_OUT + 255) / 256, BATCH / 64), 256, 0, stream>>>(
        xn, Wcov, bcov, y, BATCH, N_COV_OUT, N_FEAT);
    gemm_wmma_kernel<8, 1, 1, false><<<dim3(1, BATCH / 256), 256, 0, stream>>>(
        xn, Wlin, blin, rets, BATCH, N_ASSETS, N_FEAT);
  }

  pgd_kernel<<<BATCH, 32, 0, stream>>>(y, rets, gsq, alp, out);
}
